// VQVAE_49460843381151
// MI455X (gfx1250) — matrix operand order repaired and checked by reading, not compiled, not run on hardware
//
#include <hip/hip_runtime.h>

// ---------------------------------------------------------------------------
// VQ-VAE-2 forward for MI455X (gfx1250, wave32, WMMA).
// Implicit-GEMM convs (incl. transposed via lhs-dilation gather) using
// v_wmma_f32_16x16x32_bf16, f32 accumulate; BN/ReLU/residual fused.
// Templated on (KH,KW,STRIDE,LDIL) -> div-free im2col decode.
// WG tile 64(M) x 128(N): 8 waves (2x4), each wave owns 32x32 D
// (2 A-frags x 2 B-frags -> 4 WMMAs per K-chunk, full fragment reuse).
// ---------------------------------------------------------------------------

typedef __attribute__((ext_vector_type(16))) __bf16         v16bf;
typedef __attribute__((ext_vector_type(8)))  float          v8f;
typedef __attribute__((ext_vector_type(8)))  unsigned short v8us;
typedef __attribute__((ext_vector_type(16))) unsigned short v16us;

#define BN_SCALE 0.9999950000374997f /* 1/sqrt(1+1e-5) */

__device__ __forceinline__ unsigned short f2bf_bits(float x) {
  union { float f; unsigned u; } v; v.f = x;
  unsigned r = v.u + 0x7fffu + ((v.u >> 16) & 1u); // round-to-nearest-even
  return (unsigned short)(r >> 16);
}

__device__ __forceinline__ v16bf lds_frag(const unsigned short* p0,
                                          const unsigned short* p1) {
  v8us lo = *(const v8us*)p0;
  v8us hi = *(const v8us*)p1;
  v16us v;
  #pragma unroll
  for (int i = 0; i < 8; ++i) { v[i] = lo[i]; v[8 + i] = hi[i]; }
  return __builtin_bit_cast(v16bf, v);
}

template <int KH, int KW, int STRIDE, int LDIL>
__global__ __launch_bounds__(256) void conv_wmma(
    const float* __restrict__ in, const float* __restrict__ w,
    const float* __restrict__ bias, const float* __restrict__ bng,
    const float* __restrict__ bnb, const float* __restrict__ res,
    float* __restrict__ out,
    int N, int Cin, int Hin, int Win, int Cout, int Hout, int Wout,
    int pad, int relu)
{
  constexpr int KHW = KH * KW;
  const int LDK = 40;                                    // padded bf16 row stride
  __shared__ __align__(16) unsigned short sA[64 * LDK];  // 64 co rows  x 32 k
  __shared__ __align__(16) unsigned short sB[128 * LDK]; // 128 pos rows x 32 k

  const int tid  = threadIdx.x;
  const int lane = tid & 31;
  const int wv   = tid >> 5;
  const int wm   = wv >> 2;                 // 0..1  (M half of 64)
  const int wn   = wv & 3;                  // 0..3  (N quarter of 128)
  const unsigned HWo = (unsigned)(Hout * Wout);
  const unsigned Npos = (unsigned)N * HWo;
  const int m_base = blockIdx.y * 64;
  const unsigned n_base = (unsigned)blockIdx.x * 128u;
  const int Ktot = Cin * KHW;

  // ---- B-loader coords: 128 positions x 32 k; thread owns 16 k of one pos ----
  const int posl = tid & 127;
  const unsigned pos = n_base + (unsigned)posl;
  const bool posok = pos < Npos;
  int bn_ = 0, bho = 0, bwo = 0;
  if (posok) {
    unsigned hw = pos % HWo;
    bn_ = (int)(pos / HWo);
    bho = (int)(hw / (unsigned)Wout);
    bwo = (int)(hw % (unsigned)Wout);
  }
  const int kgrp16 = (tid >> 7) * 16;       // 0 or 16

  // ---- A-loader coords: one co row, 8 consecutive k ----
  const int arow = tid >> 2;                // 0..63
  const int akq  = (tid & 3) * 8;           // 0,8,16,24
  const int aco  = m_base + arow;
  const bool acook = aco < Cout;

  v8f acc00 = {0.f,0.f,0.f,0.f,0.f,0.f,0.f,0.f};
  v8f acc01 = {0.f,0.f,0.f,0.f,0.f,0.f,0.f,0.f};
  v8f acc10 = {0.f,0.f,0.f,0.f,0.f,0.f,0.f,0.f};
  v8f acc11 = {0.f,0.f,0.f,0.f,0.f,0.f,0.f,0.f};

  for (int kk = 0; kk < Ktot; kk += 32) {
    // ---- stage A (weights, contiguous in k) ----
    {
      const float* wp = w + (long)aco * Ktot;
      v8us t;
      #pragma unroll
      for (int j = 0; j < 8; ++j) {
        int k = kk + akq + j;
        float v = (acook && k < Ktot) ? wp[k] : 0.f;
        t[j] = f2bf_bits(v);
      }
      *(v8us*)&sA[arow * LDK + akq] = t;
    }
    // ---- stage B (im2col gather; compile-time KH/KW -> div-free) ----
    #pragma unroll
    for (int g = 0; g < 2; ++g) {
      v8us t;
      #pragma unroll
      for (int j = 0; j < 8; ++j) {
        int k = kk + kgrp16 + g * 8 + j;
        float v = 0.f;
        if (posok && k < Ktot) {
          int ci  = k / KHW;
          int r   = k - ci * KHW;
          int kh  = r / KW;
          int kw_ = r - kh * KW;
          int hp = bho * STRIDE - pad + kh;
          int wp = bwo * STRIDE - pad + kw_;
          bool ok; int hi, wi;
          if (LDIL > 1) {
            ok = (hp >= 0) && (wp >= 0) && ((hp % LDIL) == 0) && ((wp % LDIL) == 0);
            hi = hp / LDIL; wi = wp / LDIL;
            ok = ok && (hi < Hin) && (wi < Win);
          } else {
            ok = (hp >= 0) && (wp >= 0) && (hp < Hin) && (wp < Win);
            hi = hp; wi = wp;
          }
          if (ok) v = in[(((long)bn_ * Cin + ci) * Hin + hi) * Win + wi];
        }
        t[j] = f2bf_bits(v);
      }
      *(v8us*)&sB[posl * LDK + kgrp16 + g * 8] = t;
    }
    __syncthreads();

    // ---- fragments (contiguous k order, k = 16h + s) + 4 WMMAs ----
    {
      const int l16 = lane & 15;
      const int kbA = (lane < 16) ? 0 : 16;  // A: lanes hi-half hold K 16..31
      const int kbB = (lane < 16) ? 0 : 16;  // B: lanes hi-half hold K 16..31
      const unsigned short* pa0 = &sA[(wm * 32 + l16) * LDK + kbA];
      const unsigned short* pa1 = &sA[(wm * 32 + 16 + l16) * LDK + kbA];
      const unsigned short* pb0 = &sB[(wn * 32 + l16) * LDK + kbB];
      const unsigned short* pb1 = &sB[(wn * 32 + 16 + l16) * LDK + kbB];
      v16bf A0 = lds_frag(pa0, pa0 + 8);
      v16bf A1 = lds_frag(pa1, pa1 + 8);
      v16bf B0 = lds_frag(pb0, pb0 + 8);
      v16bf B1 = lds_frag(pb1, pb1 + 8);
      acc00 = __builtin_amdgcn_wmma_f32_16x16x32_bf16(false, A0, false, B0,
                                                      (short)0, acc00, false, false);
      acc01 = __builtin_amdgcn_wmma_f32_16x16x32_bf16(false, A0, false, B1,
                                                      (short)0, acc01, false, false);
      acc10 = __builtin_amdgcn_wmma_f32_16x16x32_bf16(false, A1, false, B0,
                                                      (short)0, acc10, false, false);
      acc11 = __builtin_amdgcn_wmma_f32_16x16x32_bf16(false, A1, false, B1,
                                                      (short)0, acc11, false, false);
    }
    __syncthreads();
  }

  // ---- epilogue: bias + BN + residual + ReLU ----
  const int l16 = lane & 15;
  const int mrow0 = wm * 32 + ((lane < 16) ? 0 : 8);
  #pragma unroll
  for (int nh = 0; nh < 2; ++nh) {
    const unsigned npos = n_base + (unsigned)(wn * 32 + nh * 16 + l16);
    if (npos < Npos) {
      unsigned hw = npos % HWo;
      int n = (int)(npos / HWo);
      #pragma unroll
      for (int r = 0; r < 8; ++r) {
        int co = m_base + mrow0 + r;
        float v0 = nh ? acc01[r] : acc00[r];
        float v1 = nh ? acc11[r] : acc10[r];
        if (co < Cout) {
          float y = v0 + bias[co];
          if (bng) y = y * (bng[co] * BN_SCALE) + bnb[co];
          long oi = ((long)n * Cout + co) * HWo + hw;
          if (res) y += res[oi];
          if (relu) y = (y > 0.f) ? y : 0.f;
          out[oi] = y;
        }
        int co1 = co + 16;
        if (co1 < Cout) {
          float y = v1 + bias[co1];
          if (bng) y = y * (bng[co1] * BN_SCALE) + bnb[co1];
          long oi = ((long)n * Cout + co1) * HWo + hw;
          if (res) y += res[oi];
          if (relu) y = (y > 0.f) ? y : 0.f;
          out[oi] = y;
        }
      }
    }
  }
}

// ---------------------------------------------------------------------------
// Vector quantization: one thread per spatial position (NCHW in/out).
// ---------------------------------------------------------------------------
__global__ __launch_bounds__(256) void vq_kernel(
    const float* __restrict__ ze, const float* __restrict__ Wc,
    float* __restrict__ zq, int* __restrict__ idx, float* __restrict__ dmin,
    int B, int HW)
{
  long p = (long)blockIdx.x * blockDim.x + threadIdx.x;
  if (p >= (long)B * HW) return;
  int b  = (int)(p / HW);
  int hw = (int)(p % HW);
  const float* zp = ze + (long)b * 64 * HW + hw;
  float z[64];
  #pragma unroll
  for (int e = 0; e < 64; ++e) z[e] = zp[(long)e * HW];
  float best = 3.4e38f;
  int bi = 0;
  for (int c = 0; c < 512; ++c) {
    const float* wr = Wc + c * 64;
    float d = 0.f;
    #pragma unroll
    for (int e = 0; e < 64; ++e) { float t = z[e] - wr[e]; d = fmaf(t, t, d); }
    if (d < best) { best = d; bi = c; }
  }
  float* qp = zq + (long)b * 64 * HW + hw;
  const float* wr = Wc + bi * 64;
  #pragma unroll
  for (int e = 0; e < 64; ++e) qp[(long)e * HW] = wr[e];
  idx[p] = bi;
  dmin[p] = best;
}

// Deterministic single-block loss reduction: out = (add ? out : 0) + scale*sum
__global__ __launch_bounds__(256) void reduce_loss(
    const float* __restrict__ a, long n, float scale,
    float* __restrict__ out, int add)
{
  __shared__ float sm[256];
  float s = 0.f;
  for (long i = threadIdx.x; i < n; i += 256) s += a[i];
  sm[threadIdx.x] = s;
  __syncthreads();
  for (int o = 128; o > 0; o >>= 1) {
    if ((int)threadIdx.x < o) sm[threadIdx.x] += sm[threadIdx.x + o];
    __syncthreads();
  }
  if (threadIdx.x == 0) out[0] = (add ? out[0] : 0.f) + sm[0] * scale;
}

// Channel concat: out (N, C1+C2, HW) <- [a (N,C1,HW) | b (N,C2,HW)]
__global__ __launch_bounds__(256) void concat2(
    const float* __restrict__ a, const float* __restrict__ b,
    float* __restrict__ o, int N, int C1, int C2, int HW)
{
  long i = (long)blockIdx.x * blockDim.x + threadIdx.x;
  long total = (long)N * (C1 + C2) * HW;
  if (i >= total) return;
  int hw = (int)(i % HW);
  long t = i / HW;
  int c = (int)(t % (C1 + C2));
  int n = (int)(t / (C1 + C2));
  o[i] = (c < C1) ? a[((long)n * C1 + c) * HW + hw]
                  : b[((long)n * C2 + (c - C1)) * HW + hw];
}

// ---------------------------------------------------------------------------
// Host-side orchestration
// ---------------------------------------------------------------------------
static inline void launch_conv(hipStream_t s, const float* in, const float* w,
                               const float* b, const float* g, const float* bb,
                               const float* res, float* out,
                               int N, int Cin, int Hin, int Win,
                               int Cout, int Hout, int Wout,
                               int KH, int KW, int stride, int pad, int ldil,
                               int relu)
{
  long Npos = (long)N * Hout * Wout;
  dim3 grid((unsigned)((Npos + 127) / 128), (unsigned)((Cout + 63) / 64));
  if (ldil == 2) {
    conv_wmma<4, 4, 1, 2><<<grid, 256, 0, s>>>(in, w, b, g, bb, res, out,
        N, Cin, Hin, Win, Cout, Hout, Wout, pad, relu);
  } else if (KH == 4) {
    conv_wmma<4, 4, 2, 1><<<grid, 256, 0, s>>>(in, w, b, g, bb, res, out,
        N, Cin, Hin, Win, Cout, Hout, Wout, pad, relu);
  } else if (KH == 3) {
    conv_wmma<3, 3, 1, 1><<<grid, 256, 0, s>>>(in, w, b, g, bb, res, out,
        N, Cin, Hin, Win, Cout, Hout, Wout, pad, relu);
  } else {
    conv_wmma<1, 1, 1, 1><<<grid, 256, 0, s>>>(in, w, b, g, bb, res, out,
        N, Cin, Hin, Win, Cout, Hout, Wout, pad, relu);
  }
}

// rb param block order (sorted keys): +0 b1 +1 b2 +2 be1 +3 be2 +4 g1 +5 g2 +6 w1 +7 w2
static inline void resblock(hipStream_t s, const float* const* P, int base,
                            float* X, float* T, float* Y, int N, int H, int W)
{
  launch_conv(s, X, P[base + 6], P[base + 0], P[base + 4], P[base + 2], nullptr,
              T, N, 128, H, W, 64, H, W, 3, 3, 1, 1, 1, 1);
  launch_conv(s, T, P[base + 7], P[base + 1], P[base + 5], P[base + 3], X,
              Y, N, 64, H, W, 128, H, W, 1, 1, 1, 0, 1, 1);
}

extern "C" void kernel_launch(void* const* d_in, const int* in_sizes, int n_in,
                              void* d_out, int out_size, void* d_ws, size_t ws_size,
                              hipStream_t stream)
{
  (void)in_sizes; (void)out_size;
  if (n_in < 99) return;
  // JAX pytree flatten order (dicts sorted by key, 'params' before 'x'):
  //  0 bW | 1..8 bd{cb,cw,o1b,o1w,o2b,o2w,obb,og} | 9..24 bd.rb[0..1]
  // 25..34 be{b1,b2,c1b,c1w,c2b,c2w,c3b,c3w,g1,g2} | 35..50 be.rb[0..1]
  // 51 bqc_b 52 bqc_w 53 tW | 54..57 td{cb,cw,ob,ow} | 58..73 td.rb[0..1]
  // 74..77 te{c1b,c1w,c2b,c2w} | 78..93 te.rb[0..1]
  // 94 tqc_b 95 tqc_w 96 up_b 97 up_w | 98 x
  const float* P[99];
  for (int i = 0; i < 99; ++i) P[i] = (const float*)d_in[i];

  // workspace arena (floats)
  if (ws_size < (size_t)65093632 * 4) return;
  float* ws = (float*)d_ws;
  float* RA = ws + 0;          // 16.78M : be.c1 out -> concat1 -> o1 out
  float* RB = ws + 16777216;   //  8.39M : 64x64 128ch ping
  float* RC = ws + 25165824;   //  8.39M : 64x64 128ch pong (b_enc persists)
  float* RD = ws + 33554432;   //  4.19M : 64x64 64ch temp / bq_pre
  float* RE = ws + 37748736;   //  8.39M : concat2 (ups|bq)
  float* U0 = ws + 46137344;   //  2.10M : 32x32 128ch ping
  float* U1 = ws + 48234496;   //  2.10M : 32x32 128ch pong
  float* U2 = ws + 50331648;   //  1.05M : 32x32 64ch temp
  float* TQ = ws + 51380224;   //  1.05M : quantized top (persists)
  float* TD = ws + 52428800;   //  4.19M : t_dec
  float* BQ = ws + 56623104;   //  4.19M : quantized bottom
  float* UP = ws + 60817408;   //  4.19M : upsampled top
  float* DT = ws + 65011712;   // 16384  : top min-dists
  float* DB = ws + 65028096;   // 65536  : bottom min-dists

  float* out_f  = (float*)d_out;            // out  : [0, 3145728)
  float* loss_p = out_f + 3145728;          // loss : [3145728]
  int*   tidx_p = (int*)(out_f + 3145729);  // tidx : 16384
  int*   bidx_p = (int*)(out_f + 3162113);  // bidx : 65536

  const float* x = P[98];
  const int N = 16;
  hipStream_t s = stream;

  // ---- bottom encoder (256 -> 64) ----
  launch_conv(s, x,  P[28], P[27], P[33], P[25], nullptr, RA,
              N, 3, 256, 256, 64, 128, 128, 4, 4, 2, 1, 1, 1);     // be.c1+bn+relu
  launch_conv(s, RA, P[30], P[29], P[34], P[26], nullptr, RB,
              N, 64, 128, 128, 128, 64, 64, 4, 4, 2, 1, 1, 1);     // be.c2+bn+relu
  launch_conv(s, RB, P[32], P[31], nullptr, nullptr, nullptr, RC,
              N, 128, 64, 64, 128, 64, 64, 3, 3, 1, 1, 1, 1);      // be.c3+relu
  resblock(s, P, 35, RC, RD, RB, N, 64, 64);
  resblock(s, P, 43, RB, RD, RC, N, 64, 64);                       // b_enc = RC

  // ---- top encoder (64 -> 32) ----
  launch_conv(s, RC, P[75], P[74], nullptr, nullptr, nullptr, U2,
              N, 128, 64, 64, 64, 32, 32, 4, 4, 2, 1, 1, 1);       // te.c1+relu
  launch_conv(s, U2, P[77], P[76], nullptr, nullptr, nullptr, U0,
              N, 64, 32, 32, 128, 32, 32, 3, 3, 1, 1, 1, 1);       // te.c2+relu
  resblock(s, P, 78, U0, U2, U1, N, 32, 32);
  resblock(s, P, 86, U1, U2, U0, N, 32, 32);                       // t = U0
  launch_conv(s, U0, P[95], P[94], nullptr, nullptr, nullptr, U2,
              N, 128, 32, 32, 64, 32, 32, 1, 1, 1, 0, 1, 0);       // tqc

  // ---- top VQ ----
  vq_kernel<<<(N * 1024 + 255) / 256, 256, 0, s>>>(U2, P[53], TQ, tidx_p, DT, N, 1024);
  reduce_loss<<<1, 256, 0, s>>>(DT, (long)N * 1024, 1.f / ((float)N * 1024 * 64), loss_p, 0);

  // ---- top decoder (32 -> 64) ----
  launch_conv(s, TQ, P[55], P[54], nullptr, nullptr, nullptr, U1,
              N, 64, 32, 32, 128, 32, 32, 3, 3, 1, 1, 1, 1);       // td.c+relu
  resblock(s, P, 58, U1, U2, U0, N, 32, 32);
  resblock(s, P, 66, U0, U2, U1, N, 32, 32);                       // d = U1
  launch_conv(s, U1, P[57], P[56], nullptr, nullptr, nullptr, TD,
              N, 128, 32, 32, 64, 64, 64, 4, 4, 1, 2, 2, 0);       // td.ow (convT)

  // ---- bottom quantize path ----
  { long tot = (long)N * 192 * 4096;
    concat2<<<(unsigned)((tot + 255) / 256), 256, 0, s>>>(TD, RC, RA, N, 64, 128, 4096); }
  launch_conv(s, RA, P[52], P[51], nullptr, nullptr, nullptr, RD,
              N, 192, 64, 64, 64, 64, 64, 1, 1, 1, 0, 1, 0);       // bqc
  vq_kernel<<<(N * 4096 + 255) / 256, 256, 0, s>>>(RD, P[0], BQ, bidx_p, DB, N, 4096);
  reduce_loss<<<1, 256, 0, s>>>(DB, (long)N * 4096, 1.f / ((float)N * 4096 * 64), loss_p, 1);

  // ---- bottom decoder (64 -> 256) ----
  launch_conv(s, TQ, P[97], P[96], nullptr, nullptr, nullptr, UP,
              N, 64, 32, 32, 64, 64, 64, 4, 4, 1, 2, 2, 0);        // up (convT)
  { long tot = (long)N * 128 * 4096;
    concat2<<<(unsigned)((tot + 255) / 256), 256, 0, s>>>(UP, BQ, RE, N, 64, 64, 4096); }
  launch_conv(s, RE, P[2], P[1], nullptr, nullptr, nullptr, RB,
              N, 128, 64, 64, 128, 64, 64, 3, 3, 1, 1, 1, 1);      // bd.cw+relu
  resblock(s, P, 9,  RB, RD, RC, N, 64, 64);
  resblock(s, P, 17, RC, RD, RB, N, 64, 64);                       // g = RB
  launch_conv(s, RB, P[4], P[3], P[8], P[7], nullptr, RA,
              N, 128, 64, 64, 64, 128, 128, 4, 4, 1, 2, 2, 1);     // bd.o1 (convT)+bn+relu
  launch_conv(s, RA, P[6], P[5], nullptr, nullptr, nullptr, out_f,
              N, 64, 128, 128, 3, 256, 256, 4, 4, 1, 2, 2, 0);     // bd.o2 (convT) -> out
}
